// EnsemblePredictionModule_8486855377288
// MI455X (gfx1250) — matrix operand order repaired and checked by reading, not compiled, not run on hardware
//
#include <hip/hip_runtime.h>
#include <hip/hip_bf16.h>
#include <stdint.h>

// ---------------------------------------------------------------------------
// Routed ensemble MLP for MI455X (gfx1250, wave32, WMMA).
//
// B=16384, D=1024, H=1024, O=256, C=8, L=3.
// Dense reference = ~0.9 TFLOP; routing (compute only the argmax branch per
// sample) = ~0.11 TFLOP. HBM traffic ~350MB (incl. one-time bf16 repack)
// ~15us @ 23.3TB/s => compute bound => v_wmma_f32_16x16x32_bf16 GEMMs with
// double-buffered LDS fed by global_load_async_to_lds_b128 (ASYNCcnt).
// ---------------------------------------------------------------------------

#define BDIM 16384
#define DDIM 1024
#define HDIM 1024
#define ODIM 256
#define CDIM 8

typedef __bf16 v16bf __attribute__((ext_vector_type(16)));
typedef float  v8f   __attribute__((ext_vector_type(8)));

#define AST 48   // LDS row stride (elements): 96B rows, all fragment chunks 16B-aligned

// --- CDNA5 async global->LDS copy (VGLOBAL, tracked by ASYNCcnt) ---
__device__ __forceinline__ void async_ld_b128(uint32_t ldsAddr, const void* gaddr) {
  asm volatile("global_load_async_to_lds_b128 %0, %1, off"
               :: "v"(ldsAddr), "v"(gaddr) : "memory");
}
__device__ __forceinline__ void wait_asynccnt0() {
  asm volatile("s_wait_asynccnt 0x0" ::: "memory");
}

// ---------------------------------------------------------------------------
// Kernel 1: classifier logits + argmax routing. One wave (32 lanes) per sample.
// ---------------------------------------------------------------------------
__global__ __launch_bounds__(256) void k_logits(const float* __restrict__ x,
                                                const float* __restrict__ Wc,
                                                const float* __restrict__ bc,
                                                float* __restrict__ logits,
                                                int* __restrict__ cls) {
  __shared__ float sWc[DDIM * CDIM];  // 32KB: whole classifier weight
  for (int i = threadIdx.x; i < DDIM * CDIM; i += 256) sWc[i] = Wc[i];
  __syncthreads();

  const int wave = threadIdx.x >> 5;
  const int lane = threadIdx.x & 31;
  const int s = blockIdx.x * 8 + wave;
  const float* xr = x + (size_t)s * DDIM;

  float acc[CDIM];
#pragma unroll
  for (int c = 0; c < CDIM; ++c) acc[c] = 0.0f;

  for (int k = lane; k < DDIM; k += 32) {
    const float xv = xr[k];
    const float* w = &sWc[k * CDIM];
#pragma unroll
    for (int c = 0; c < CDIM; ++c) acc[c] += xv * w[c];
  }
#pragma unroll
  for (int off = 16; off >= 1; off >>= 1) {
#pragma unroll
    for (int c = 0; c < CDIM; ++c) acc[c] += __shfl_xor(acc[c], off, 32);
  }
#pragma unroll
  for (int c = 0; c < CDIM; ++c) acc[c] += bc[c];

  if (lane < CDIM) logits[(size_t)s * CDIM + lane] = acc[lane];
  if (lane == 0) {
    int best = 0;
    float bv = acc[0];
#pragma unroll
    for (int c = 1; c < CDIM; ++c) {
      if (acc[c] > bv) { bv = acc[c]; best = c; }  // first-max, like jnp.argmax
    }
    cls[s] = best;
  }
}

// ---------------------------------------------------------------------------
// Routing bookkeeping
// ---------------------------------------------------------------------------
__global__ void k_zero(int* __restrict__ counts) {
  if (threadIdx.x < CDIM) counts[threadIdx.x] = 0;
}

__global__ __launch_bounds__(256) void k_compact(const int* __restrict__ cls,
                                                 int* __restrict__ counts,
                                                 int* __restrict__ perm) {
  const int i = blockIdx.x * 256 + threadIdx.x;
  const int c = cls[i];
  const int pos = atomicAdd(&counts[c], 1);
  perm[c * BDIM + pos] = i;
}

__global__ void k_offsets(const int* __restrict__ counts, int* __restrict__ offs) {
  if (threadIdx.x == 0) {
    int o = 0;
#pragma unroll
    for (int c = 0; c < CDIM; ++c) { offs[c] = o; o += counts[c]; }
  }
}

// ---------------------------------------------------------------------------
// One-time repack: x fp32 -> bf16
// ---------------------------------------------------------------------------
__global__ __launch_bounds__(256) void k_cvt_x(const float* __restrict__ in,
                                               __bf16* __restrict__ out) {
  const size_t i = ((size_t)blockIdx.x * 256 + threadIdx.x) * 8;
  const float4 a = *(const float4*)(in + i);
  const float4 b = *(const float4*)(in + i + 4);
  union { __bf16 e[8]; float4 q; } u;
  u.e[0] = (__bf16)a.x; u.e[1] = (__bf16)a.y; u.e[2] = (__bf16)a.z; u.e[3] = (__bf16)a.w;
  u.e[4] = (__bf16)b.x; u.e[5] = (__bf16)b.y; u.e[6] = (__bf16)b.z; u.e[7] = (__bf16)b.w;
  *(float4*)(out + i) = u.q;
}

// ---------------------------------------------------------------------------
// One-time repack: weights (K x N fp32, row-major) -> (N x K bf16, row-major)
// ---------------------------------------------------------------------------
__global__ __launch_bounds__(256) void k_transpose(const float* __restrict__ in,
                                                   __bf16* __restrict__ out,
                                                   int K, int N,
                                                   long inMatStride, long outMatStride) {
  __shared__ float t[32][33];
  const float* src = in + (size_t)blockIdx.z * inMatStride;
  __bf16* dst = out + (size_t)blockIdx.z * outMatStride;
  const int n0 = blockIdx.x * 32;
  const int k0 = blockIdx.y * 32;
  const int tid = threadIdx.x;
  {
    const int kr = tid >> 3;         // 0..31
    const int nc = (tid & 7) * 4;    // 0..28
    const float4 v = *(const float4*)(src + (size_t)(k0 + kr) * N + n0 + nc);
    t[kr][nc + 0] = v.x; t[kr][nc + 1] = v.y; t[kr][nc + 2] = v.z; t[kr][nc + 3] = v.w;
  }
  __syncthreads();
  {
    const int nr = tid >> 3;         // 0..31
    const int kc = (tid & 7) * 4;    // 0..28
    union { __bf16 e[4]; float2 q; } u;
#pragma unroll
    for (int j = 0; j < 4; ++j) u.e[j] = (__bf16)t[kc + j][nr];
    *(float2*)(dst + (size_t)(n0 + nr) * K + k0 + kc) = u.q;
  }
}

// ---------------------------------------------------------------------------
// WMMA GEMM: per class c, rows [0,counts[c]) of compacted samples:
//   out = act( A(rows x K) @ W_c(N x K)^T + b_c )
// Block = 256 threads = 8 waves; block tile 128(M) x 128(N); K-step 32.
// Wave tile 32x64 => 2x4 fragments of v_wmma_f32_16x16x32_bf16.
// Double-buffered LDS fed by async global->LDS DMA (no VGPR staging).
// GATHER    : A rows gathered from bf16 xb via perm
// FINAL_OUT : scatter fp32 rows (+bias, no relu) to d_out via perm
// ---------------------------------------------------------------------------
template <bool GATHER, bool FINAL_OUT>
__global__ __launch_bounds__(256, 1) void k_gemm(
    const __bf16* __restrict__ Ain,        // xb (GATHER) or compacted bf16 acts
    const __bf16* __restrict__ Wt,         // per-class (N x K) bf16
    long wStride,
    const float* __restrict__ bbase, long bStride,
    const int* __restrict__ counts, const int* __restrict__ offs,
    const int* __restrict__ perm,
    __bf16* __restrict__ outB,             // compacted bf16 out (relu)
    float* __restrict__ outF,              // d_out (FINAL_OUT)
    int K, int N) {
  const int c = blockIdx.z;
  const int cnt = counts[c];
  const int m0 = blockIdx.y * 128;
  if (m0 >= cnt) return;  // routed: skip tiles beyond this class's population
  const int n0 = blockIdx.x * 128;

  const __bf16* W = Wt + (long)c * wStride;      // N x K row-major (K contiguous)
  const float* bias = bbase + (long)c * bStride; // N
  const int off = offs[c];

  __shared__ __bf16 As[2][128 * AST];
  __shared__ __bf16 Bs[2][128 * AST];
  const uint32_t bufBytes = 128 * AST * 2;       // 12KB per buffer

  const int tid = threadIdx.x;
  const int lane = tid & 31;
  const int wave = tid >> 5;
  const int wm = wave & 3;   // M quadrant (32 rows)
  const int wn = wave >> 2;  // N half (64 cols)
  const int l16 = lane & 15;
  const int half = lane >> 4;

  // --- staging: 16B (8 bf16) per thread, 64 rows per pass, 2 passes ---
  const int rowInPass = tid >> 2;        // 0..63
  const int kOff = (tid & 3) * 8;        // 0,8,16,24 (elements)
  const __bf16* aPtr[2];
  const __bf16* bPtr[2];
  uint32_t aLds[2], bLds[2];             // LDS byte addresses, buffer 0
#pragma unroll
  for (int p = 0; p < 2; ++p) {
    const int rl = m0 + p * 64 + rowInPass;
    const int rc = (rl < cnt) ? rl : (cnt - 1);
    if constexpr (GATHER) {
      aPtr[p] = Ain + (size_t)perm[c * BDIM + rc] * K + kOff;
    } else {
      aPtr[p] = Ain + (size_t)(off + rc) * K + kOff;
    }
    bPtr[p] = W + (size_t)(n0 + p * 64 + rowInPass) * K + kOff;
    const int lo = (p * 64 + rowInPass) * AST + kOff;
    aLds[p] = (uint32_t)(uintptr_t)&As[0][lo];
    bLds[p] = (uint32_t)(uintptr_t)&Bs[0][lo];
  }

  v8f acc[2][4] = {};

  // prologue: async-stage K-step 0 into buffer 0
#pragma unroll
  for (int p = 0; p < 2; ++p) {
    async_ld_b128(aLds[p], aPtr[p]);
    async_ld_b128(bLds[p], bPtr[p]);
  }
  wait_asynccnt0();
  __syncthreads();

  const int S = K / 32;
  for (int s = 0; s < S; ++s) {
    const int cur = s & 1;
    // async-stage next K-step into the other buffer while WMMAs run
    if (s + 1 < S) {
      const int kn = (s + 1) * 32;
      const uint32_t bo = (uint32_t)(cur ^ 1) * bufBytes;
#pragma unroll
      for (int p = 0; p < 2; ++p) {
        async_ld_b128(aLds[p] + bo, aPtr[p] + kn);
        async_ld_b128(bLds[p] + bo, bPtr[p] + kn);
      }
    }
    if (s + 2 < S) {  // L2 prefetch hints two steps out
      const int kp = (s + 2) * 32;
      __builtin_prefetch(aPtr[0] + kp, 0, 1);
      __builtin_prefetch(bPtr[0] + kp, 0, 1);
    }

    // ---- fragments per ISA 16-bit layouts, 8 WMMAs ----
    union Frag { v16bf v; float4 q[2]; };
    Frag a[2], b[4];
#pragma unroll
    for (int fm = 0; fm < 2; ++fm) {
      // A 16x32: lanes 0-15 row M, K chunks [half*8,+8) and [half*8+16,+8)
      const __bf16* ap = &As[cur][(wm * 32 + fm * 16 + l16) * AST + half * 8];
      a[fm].q[0] = *(const float4*)(ap);
      a[fm].q[1] = *(const float4*)(ap + 16);
    }
#pragma unroll
    for (int fn = 0; fn < 4; ++fn) {
      // B 32x16: lane = column, K chunks [half*8,+8) and [half*8+16,+8)
      const __bf16* bp = &Bs[cur][(wn * 64 + fn * 16 + l16) * AST + half * 8];
      b[fn].q[0] = *(const float4*)(bp);
      b[fn].q[1] = *(const float4*)(bp + 16);
    }
#pragma unroll
    for (int fm = 0; fm < 2; ++fm)
#pragma unroll
      for (int fn = 0; fn < 4; ++fn)
        acc[fm][fn] = __builtin_amdgcn_wmma_f32_16x16x32_bf16(
            false, a[fm].v, false, b[fn].v, (short)0, acc[fm][fn], false, false);

    if (s + 1 < S) wait_asynccnt0();  // own async writes to the other buffer done
    __syncthreads();                  // cross-wave: buffers consistent
  }

  // ---- epilogue: C/D layout => lane l16 = N col, VGPR r = M row (+8 if lane>=16)
  const int rAdd = half * 8;
  float bv[4];
#pragma unroll
  for (int fn = 0; fn < 4; ++fn) bv[fn] = bias[n0 + wn * 64 + fn * 16 + l16];
  const int colBase = n0 + wn * 64 + l16;
  const bool full = (m0 + 128 <= cnt);

#pragma unroll
  for (int fm = 0; fm < 2; ++fm) {
#pragma unroll
    for (int r = 0; r < 8; ++r) {
      const int rowLocal = m0 + wm * 32 + fm * 16 + r + rAdd;
      if (!full && rowLocal >= cnt) continue;
      if constexpr (FINAL_OUT) {
        float* po = outF + (size_t)perm[c * BDIM + rowLocal] * ODIM + colBase;
#pragma unroll
        for (int fn = 0; fn < 4; ++fn) po[fn * 16] = acc[fm][fn][r] + bv[fn];
      } else {
        __bf16* po = outB + (size_t)(off + rowLocal) * N + colBase;
#pragma unroll
        for (int fn = 0; fn < 4; ++fn) {
          float v = acc[fm][fn][r] + bv[fn];
          v = v > 0.0f ? v : 0.0f;  // relu
          po[fn * 16] = (__bf16)v;
        }
      }
    }
  }
}

// ---------------------------------------------------------------------------
extern "C" void kernel_launch(void* const* d_in, const int* in_sizes, int n_in,
                              void* d_out, int out_size, void* d_ws, size_t ws_size,
                              hipStream_t stream) {
  const float* x  = (const float*)d_in[0];
  const float* Wc = (const float*)d_in[1];
  const float* bc = (const float*)d_in[2];
  const float* W1 = (const float*)d_in[3];
  const float* b1 = (const float*)d_in[4];
  const float* W2 = (const float*)d_in[5];
  const float* b2 = (const float*)d_in[6];
  const float* Wo = (const float*)d_in[7];
  const float* bo = (const float*)d_in[8];

  float* out    = (float*)d_out;                 // (B, O)
  float* logits = out + (size_t)BDIM * ODIM;     // (B, C)

  // workspace layout (~149 MB, all 256B-aligned)
  char* w = (char*)d_ws;
  int* cls    = (int*)w;     w += (size_t)BDIM * 4;
  int* counts = (int*)w;     w += 256;
  int* offs   = (int*)w;     w += 256;
  int* perm   = (int*)w;     w += (size_t)CDIM * BDIM * 4;
  __bf16* xb  = (__bf16*)w;  w += (size_t)BDIM * DDIM * 2;
  __bf16* Wt1 = (__bf16*)w;  w += (size_t)CDIM * HDIM * DDIM * 2;
  __bf16* Wt2 = (__bf16*)w;  w += (size_t)CDIM * 2 * HDIM * HDIM * 2;
  __bf16* Wto = (__bf16*)w;  w += (size_t)CDIM * ODIM * HDIM * 2;
  __bf16* hA  = (__bf16*)w;  w += (size_t)BDIM * HDIM * 2;
  __bf16* hB  = (__bf16*)w;

  // routing
  k_zero<<<1, 32, 0, stream>>>(counts);
  k_logits<<<BDIM / 8, 256, 0, stream>>>(x, Wc, bc, logits, cls);
  k_compact<<<BDIM / 256, 256, 0, stream>>>(cls, counts, perm);
  k_offsets<<<1, 32, 0, stream>>>(counts, offs);

  // one-time bf16 repack (x) and repack+transpose (weights)
  k_cvt_x<<<(int)(((size_t)BDIM * DDIM) / (256 * 8)), 256, 0, stream>>>(x, xb);
  k_transpose<<<dim3(HDIM / 32, DDIM / 32, CDIM), 256, 0, stream>>>(
      W1, Wt1, DDIM, HDIM, (long)DDIM * HDIM, (long)HDIM * DDIM);
  k_transpose<<<dim3(HDIM / 32, HDIM / 32, CDIM * 2), 256, 0, stream>>>(
      W2, Wt2, HDIM, HDIM, (long)HDIM * HDIM, (long)HDIM * HDIM);
  k_transpose<<<dim3(ODIM / 32, HDIM / 32, CDIM), 256, 0, stream>>>(
      Wo, Wto, HDIM, ODIM, (long)HDIM * ODIM, (long)ODIM * HDIM);

  const dim3 blk(256);
  const dim3 gH(HDIM / 128, BDIM / 128, CDIM);
  const dim3 gO(ODIM / 128, BDIM / 128, CDIM);

  // layer 1: gather xb @ W1[c] -> hA (relu, bf16)
  k_gemm<true, false><<<gH, blk, 0, stream>>>(
      xb, Wt1, (long)HDIM * DDIM, b1, (long)HDIM,
      counts, offs, perm, hA, nullptr, DDIM, HDIM);
  // layer 2: hA @ W2[c,0] -> hB
  k_gemm<false, false><<<gH, blk, 0, stream>>>(
      hA, Wt2, (long)2 * HDIM * HDIM, b2, (long)2 * HDIM,
      counts, offs, perm, hB, nullptr, HDIM, HDIM);
  // layer 3: hB @ W2[c,1] -> hA
  k_gemm<false, false><<<gH, blk, 0, stream>>>(
      hB, Wt2 + (size_t)HDIM * HDIM, (long)2 * HDIM * HDIM,
      b2 + HDIM, (long)2 * HDIM,
      counts, offs, perm, hA, nullptr, HDIM, HDIM);
  // layer 4: hA @ Wo[c] + bo -> scatter fp32 rows to d_out
  k_gemm<false, true><<<gO, blk, 0, stream>>>(
      hA, Wto, (long)ODIM * HDIM, bo, (long)ODIM,
      counts, offs, perm, nullptr, out, HDIM, ODIM);
}
